// PowerFlowSoftGNN_12678743458342
// MI455X (gfx1250) — compile-verified, matrix operand order checked by reading, not run on hardware
//
#include <hip/hip_runtime.h>
#include <hip/hip_bf16.h>

// ---------------------------------------------------------------------------
// PowerFlow soft-GNN on MI455X (gfx1250, wave32, WMMA).
// Edge MLP + node update run as bf16 WMMA (f32 accumulate): memory-bound
// workload (~10 FLOP/B vs 23.3 TB/s HBM), so bf16 minimizes matrix-
// instruction issue pressure while f32 gather/scatter traffic dominates.
// ---------------------------------------------------------------------------

typedef __attribute__((ext_vector_type(16))) __bf16 bf16x16;
typedef __attribute__((ext_vector_type(8)))  float   f32x8;
typedef __attribute__((ext_vector_type(4)))  float   f32x4;
typedef __attribute__((ext_vector_type(4)))  int     i32x4;
typedef __attribute__((ext_vector_type(4)))  unsigned int u32x4;
typedef __attribute__((ext_vector_type(2)))  unsigned int u32x2;

struct U32x8 { u32x4 lo, hi; };

#define XSTR 104   // padded row stride in bf16 elements (52 dwords, bank-conflict free)

__device__ __forceinline__ unsigned short f2bf(float f) {
  unsigned int u = __float_as_uint(f);
  unsigned int r = (u + 0x7FFFu + ((u >> 16) & 1u)) >> 16;   // round-to-nearest-even
  return (unsigned short)r;
}
__device__ __forceinline__ unsigned int pack2(float a, float b) {
  return (unsigned int)f2bf(a) | ((unsigned int)f2bf(b) << 16);
}
// Build a 16x(bf16) A/B fragment from two 16B LDS chunks (K {k0..k0+7, k0+16..k0+23})
__device__ __forceinline__ bf16x16 mk_frag(const unsigned short* p) {
  U32x8 r;
  r.lo = *(const u32x4*)p;
  r.hi = *(const u32x4*)(p + 16);
  return __builtin_bit_cast(bf16x16, r);
}

// ---------------------------------------------------------------------------
// Kernel 1: node init. vbuf[n] = [1, 0, (PQ @ W_in + b_in)(32), 0, 0]
// ---------------------------------------------------------------------------
__global__ __launch_bounds__(256) void node_init_k(
    const float* __restrict__ PQ, const float* __restrict__ Win,
    const float* __restrict__ bin, float* __restrict__ Vpred,
    float* __restrict__ vbuf, int N)
{
  int n = blockIdx.x * 256 + threadIdx.x;
  if (n >= N) return;
  float p = PQ[2 * n], q = PQ[2 * n + 1];
  float* vb = vbuf + (size_t)n * 36;
  vb[0] = 1.0f; vb[1] = 0.0f;
  Vpred[2 * n] = 1.0f; Vpred[2 * n + 1] = 0.0f;
#pragma unroll
  for (int k = 0; k < 32; ++k)
    vb[2 + k] = fmaf(p, Win[k], fmaf(q, Win[32 + k], bin[k]));
  vb[34] = 0.0f; vb[35] = 0.0f;
}

// ---------------------------------------------------------------------------
// Kernel 2: edge message MLP (WMMA) + masked scatter-add.
// One wave per tile of 16 edges. K layout (padded to 96):
//   0..35  sender vbuf row (34 feats + 2 zeros)
//  36..71  receiver vbuf row (34 feats + 2 zeros)
//  72..73  edge features, rest zero.
// sW holds W_msg repacked as [n][k] bf16 in the same padded K layout.
// ---------------------------------------------------------------------------
__global__ __launch_bounds__(256) void edge_msg_wmma_k(
    const float* __restrict__ vbuf, const int* __restrict__ senders,
    const int* __restrict__ receivers, const float* __restrict__ ef,
    const float* __restrict__ emask, const float* __restrict__ Wmsg,
    const float* __restrict__ bmsg, float* __restrict__ agg, int E)
{
  __shared__ __align__(16) unsigned short sW[32 * XSTR];
  __shared__ __align__(16) unsigned short sX[8][16 * XSTR];
  __shared__ __align__(16) int   sSd[8][16];
  __shared__ __align__(16) int   sRv[8][16];
  __shared__ __align__(16) float sMk[8][16];

  const int tid = threadIdx.x;

  // ---- repack W_msg[70][32] into padded column-major bf16 sW[n*104 + k] ----
  for (int i = tid; i < 32 * XSTR; i += 256) {
    int n = i / XSTR, k = i - n * XSTR;
    float w = 0.0f;
    if (k < 34)                 w = Wmsg[k * 32 + n];          // sender rows 0..33
    else if (k >= 36 && k < 70) w = Wmsg[(k - 2) * 32 + n];    // receiver rows 34..67
    else if (k == 72)           w = Wmsg[68 * 32 + n];         // edge feat rows
    else if (k == 73)           w = Wmsg[69 * 32 + n];
    sW[i] = f2bf(w);
  }
  __syncthreads();

  const int lane  = tid & 31;
  const int wave  = tid >> 5;
  const int tiles = (E + 15) >> 4;
  const long t    = (long)blockIdx.x * 8 + wave;
  if (t >= tiles) return;                    // wave-uniform; EXEC all-1 inside

  const int nlo   = lane & 15;
  const int khalf = (lane >> 4) * 8;

  // ---- B fragments (weights), held in registers across the tile ----
  bf16x16 bfrag[2][3];
#pragma unroll
  for (int h = 0; h < 2; ++h)
#pragma unroll
    for (int ks = 0; ks < 3; ++ks)
      bfrag[h][ks] = mk_frag(&sW[(h * 16 + nlo) * XSTR + ks * 32 + khalf]);

  const int ebase = (int)(t << 4);
  unsigned short* X = &sX[wave][0];

  // ---- stage per-edge indices / mask / edge feats / zero tail ----
  if (lane < 16) {
    int e = ebase + lane;
    int s = 0, r = 0; float mk = 0.0f, e0 = 0.0f, e1 = 0.0f;
    if (e < E) { s = senders[e]; r = receivers[e]; mk = emask[e];
                 e0 = ef[2 * e]; e1 = ef[2 * e + 1]; }
    sSd[wave][lane] = s; sRv[wave][lane] = r; sMk[wave][lane] = mk;
    unsigned int* row32 = (unsigned int*)(X + lane * XSTR);
    row32[36] = pack2(e0, e1);               // elements 72,73
#pragma unroll
    for (int i = 37; i < 52; ++i) row32[i] = 0u;   // elements 74..103
  }
  asm volatile("s_wait_dscnt 0" ::: "memory");

  // ---- cooperative gather, division-free mapping:
  //      lane L<16  -> sender row of edge L   -> X[L][0..35]
  //      lane L>=16 -> receiver row of edge L-16 -> X[L-16][36..71]
  {
    const bool isSend = lane < 16;
    const int  gnode  = isSend ? sSd[wave][nlo] : sRv[wave][nlo];
    const float* gsrc = vbuf + (size_t)gnode * 36;
    unsigned short* gdst = X + nlo * XSTR + (isSend ? 0 : 36);
#pragma unroll
    for (int it = 0; it < 9; ++it) {
      f32x4 v = *(const f32x4*)(gsrc + it * 4);
      u32x2 pk; pk.x = pack2(v.x, v.y); pk.y = pack2(v.z, v.w);
      *(u32x2*)(gdst + it * 4) = pk;
    }
  }
  asm volatile("s_wait_dscnt 0" ::: "memory");

  // ---- 6 x v_wmma_f32_16x16x32_bf16 ----
  f32x8 acc0 = {}; f32x8 acc1 = {};
#pragma unroll
  for (int ks = 0; ks < 3; ++ks) {
    bf16x16 afrag = mk_frag(X + nlo * XSTR + ks * 32 + khalf);
    acc0 = __builtin_amdgcn_wmma_f32_16x16x32_bf16(
        false, afrag, false, bfrag[0][ks], (short)0, acc0, false, false);
    acc1 = __builtin_amdgcn_wmma_f32_16x16x32_bf16(
        false, afrag, false, bfrag[1][ks], (short)0, acc1, false, false);
  }

  // ---- epilogue: bias + relu + mask, atomic scatter to agg[recv][n] ----
  // C/D layout: lane = N%16 + 16*(M>=8); VGPR j = M%8.
  const int mhi = (lane >> 4) * 8;
  float mk[8]; int rv[8];
  {
    f32x4 a = *(const f32x4*)&sMk[wave][mhi];
    f32x4 b = *(const f32x4*)&sMk[wave][mhi + 4];
    mk[0]=a.x; mk[1]=a.y; mk[2]=a.z; mk[3]=a.w;
    mk[4]=b.x; mk[5]=b.y; mk[6]=b.z; mk[7]=b.w;
    i32x4 c = *(const i32x4*)&sRv[wave][mhi];
    i32x4 d = *(const i32x4*)&sRv[wave][mhi + 4];
    rv[0]=c.x; rv[1]=c.y; rv[2]=c.z; rv[3]=c.w;
    rv[4]=d.x; rv[5]=d.y; rv[6]=d.z; rv[7]=d.w;
  }
  const float bn0 = bmsg[nlo];
  const float bn1 = bmsg[16 + nlo];
  if (ebase + 16 <= E) {                     // full tile: wave-uniform fast path
#pragma unroll
    for (int j = 0; j < 8; ++j) {
      float* dst = agg + (size_t)rv[j] * 32;
      atomicAdd(dst + nlo,      fmaxf(acc0[j] + bn0, 0.0f) * mk[j]);
      atomicAdd(dst + 16 + nlo, fmaxf(acc1[j] + bn1, 0.0f) * mk[j]);
    }
  } else {                                   // partial last tile
#pragma unroll
    for (int j = 0; j < 8; ++j) {
      if (ebase + mhi + j < E) {
        float* dst = agg + (size_t)rv[j] * 32;
        atomicAdd(dst + nlo,      fmaxf(acc0[j] + bn0, 0.0f) * mk[j]);
        atomicAdd(dst + 16 + nlo, fmaxf(acc1[j] + bn1, 0.0f) * mk[j]);
      }
    }
  }
}

// ---------------------------------------------------------------------------
// Kernel 3: node update GEMM (WMMA): h = relu([vbuf(34), agg(32)] @ W_upd + b)
// K layout: 0..35 vbuf row (34+2 zeros), 36..67 agg row, rest zero.
// ---------------------------------------------------------------------------
__global__ __launch_bounds__(256) void node_upd_wmma_k(
    const float* __restrict__ vbuf, const float* __restrict__ agg,
    const float* __restrict__ Wupd, const float* __restrict__ bupd,
    float* __restrict__ hbuf, int N)
{
  __shared__ __align__(16) unsigned short sW[32 * XSTR];
  __shared__ __align__(16) unsigned short sX[8][16 * XSTR];

  const int tid = threadIdx.x;
  for (int i = tid; i < 32 * XSTR; i += 256) {
    int n = i / XSTR, k = i - n * XSTR;
    float w = 0.0f;
    if (k < 34)                 w = Wupd[k * 32 + n];          // v rows 0..33
    else if (k >= 36 && k < 68) w = Wupd[(k - 2) * 32 + n];    // agg rows 34..65
    sW[i] = f2bf(w);
  }
  __syncthreads();

  const int lane  = tid & 31;
  const int wave  = tid >> 5;
  const int tiles = (N + 15) >> 4;
  const long t    = (long)blockIdx.x * 8 + wave;
  if (t >= tiles) return;

  const int nlo   = lane & 15;
  const int khalf = (lane >> 4) * 8;

  bf16x16 bfrag[2][3];
#pragma unroll
  for (int h = 0; h < 2; ++h)
#pragma unroll
    for (int ks = 0; ks < 3; ++ks)
      bfrag[h][ks] = mk_frag(&sW[(h * 16 + nlo) * XSTR + ks * 32 + khalf]);

  const int nbase = (int)(t << 4);
  unsigned short* X = &sX[wave][0];

  if (lane < 16) {                           // zero tail: elements 68..103
    unsigned int* row32 = (unsigned int*)(X + lane * XSTR);
#pragma unroll
    for (int i = 34; i < 52; ++i) row32[i] = 0u;
  }

  // ---- gather, division-free mapping:
  //      lane L<16  -> vbuf row of node nbase+L  (9 float4) -> X[L][0..35]
  //      lane L>=16 -> agg  row of node nbase+L-16 (8 float4) -> X[L-16][36..67]
  {
    const int  gnode  = nbase + nlo;
    const bool valid  = gnode < N;
    if (lane < 16) {
      const float* src = vbuf + (size_t)gnode * 36;
      unsigned short* gdst = X + nlo * XSTR;
#pragma unroll
      for (int it = 0; it < 9; ++it) {
        f32x4 v = {};
        if (valid) v = *(const f32x4*)(src + it * 4);
        u32x2 pk; pk.x = pack2(v.x, v.y); pk.y = pack2(v.z, v.w);
        *(u32x2*)(gdst + it * 4) = pk;
      }
    } else {
      const float* src = agg + (size_t)gnode * 32;
      unsigned short* gdst = X + nlo * XSTR + 36;
#pragma unroll
      for (int it = 0; it < 8; ++it) {
        f32x4 v = {};
        if (valid) v = *(const f32x4*)(src + it * 4);
        u32x2 pk; pk.x = pack2(v.x, v.y); pk.y = pack2(v.z, v.w);
        *(u32x2*)(gdst + it * 4) = pk;
      }
    }
  }
  asm volatile("s_wait_dscnt 0" ::: "memory");

  f32x8 acc0 = {}; f32x8 acc1 = {};
#pragma unroll
  for (int ks = 0; ks < 3; ++ks) {
    bf16x16 afrag = mk_frag(X + nlo * XSTR + ks * 32 + khalf);
    acc0 = __builtin_amdgcn_wmma_f32_16x16x32_bf16(
        false, afrag, false, bfrag[0][ks], (short)0, acc0, false, false);
    acc1 = __builtin_amdgcn_wmma_f32_16x16x32_bf16(
        false, afrag, false, bfrag[1][ks], (short)0, acc1, false, false);
  }

  const int mhi = (lane >> 4) * 8;
  const float bn0 = bupd[nlo];
  const float bn1 = bupd[16 + nlo];
  if (nbase + 16 <= N) {                     // full tile fast path
#pragma unroll
    for (int j = 0; j < 8; ++j) {
      float* dst = hbuf + (size_t)(nbase + mhi + j) * 32;
      dst[nlo]      = fmaxf(acc0[j] + bn0, 0.0f);
      dst[16 + nlo] = fmaxf(acc1[j] + bn1, 0.0f);
    }
  } else {
#pragma unroll
    for (int j = 0; j < 8; ++j) {
      int node = nbase + mhi + j;
      if (node < N) {
        float* dst = hbuf + (size_t)node * 32;
        dst[nlo]      = fmaxf(acc0[j] + bn0, 0.0f);
        dst[16 + nlo] = fmaxf(acc1[j] + bn1, 0.0f);
      }
    }
  }
}

// ---------------------------------------------------------------------------
// Kernel 4: delta head + V update + rebuild vbuf for next layer.
// ---------------------------------------------------------------------------
__global__ __launch_bounds__(256) void node_finish_k(
    const float* __restrict__ hbuf, const float* __restrict__ Wd,
    const float* __restrict__ bd, float* __restrict__ Vpred,
    float* __restrict__ vbuf, float* __restrict__ outv, int N)
{
  int n = blockIdx.x * 256 + threadIdx.x;
  if (n >= N) return;
  const float* h = hbuf + (size_t)n * 32;
  float d0 = bd[0], d1 = bd[1];
#pragma unroll
  for (int k = 0; k < 32; ++k) {
    float hv = h[k];
    d0 = fmaf(hv, Wd[2 * k],     d0);
    d1 = fmaf(hv, Wd[2 * k + 1], d1);
  }
  float v0 = Vpred[2 * n] + d0;
  float v1 = Vpred[2 * n + 1] + d1;
  Vpred[2 * n] = v0; Vpred[2 * n + 1] = v1;
  float* vb = vbuf + (size_t)n * 36;
  vb[0] = v0; vb[1] = v1;
#pragma unroll
  for (int k = 0; k < 32; ++k) vb[2 + k] = h[k];
  vb[34] = 0.0f; vb[35] = 0.0f;
  if (outv) { outv[2 * n] = v0; outv[2 * n + 1] = v1; }
}

// ---------------------------------------------------------------------------
extern "C" void kernel_launch(void* const* d_in, const int* in_sizes, int n_in,
                              void* d_out, int out_size, void* d_ws, size_t ws_size,
                              hipStream_t stream) {
  const float* PQ       = (const float*)d_in[0];
  const int*   senders  = (const int*)  d_in[1];
  const int*   receivers= (const int*)  d_in[2];
  const float* ef       = (const float*)d_in[3];
  const float* emask    = (const float*)d_in[4];
  const float* Win      = (const float*)d_in[5];
  const float* bin      = (const float*)d_in[6];
  const float* Wmsg     = (const float*)d_in[7];
  const float* bmsg     = (const float*)d_in[8];
  const float* Wupd     = (const float*)d_in[9];
  const float* bupd     = (const float*)d_in[10];
  const float* Wd       = (const float*)d_in[11];
  const float* bd       = (const float*)d_in[12];

  const int N = in_sizes[0] / 2;
  const int E = in_sizes[1];

  // workspace carve (256B aligned)
  char* ws = (char*)d_ws;
  auto carve = [&](size_t bytes) {
    char* p = ws;
    ws += (bytes + 255) & ~(size_t)255;
    return p;
  };
  float* vbuf  = (float*)carve((size_t)N * 36 * sizeof(float));
  float* agg   = (float*)carve((size_t)N * 32 * sizeof(float));
  float* hbuf  = (float*)carve((size_t)N * 32 * sizeof(float));
  float* Vpred = (float*)carve((size_t)N * 2  * sizeof(float));

  const int nb1 = (N + 255) / 256;
  node_init_k<<<nb1, 256, 0, stream>>>(PQ, Win, bin, Vpred, vbuf, N);

  const int etiles  = (E + 15) / 16;
  const int eblocks = (etiles + 7) / 8;
  const int ntiles  = (N + 15) / 16;
  const int nblocks = (ntiles + 7) / 8;

  for (int l = 0; l < 3; ++l) {
    hipMemsetAsync(agg, 0, (size_t)N * 32 * sizeof(float), stream);
    edge_msg_wmma_k<<<eblocks, 256, 0, stream>>>(
        vbuf, senders, receivers, ef, emask,
        Wmsg + (size_t)l * 70 * 32, bmsg + (size_t)l * 32, agg, E);
    node_upd_wmma_k<<<nblocks, 256, 0, stream>>>(
        vbuf, agg, Wupd + (size_t)l * 66 * 32, bupd + (size_t)l * 32, hbuf, N);
    node_finish_k<<<nb1, 256, 0, stream>>>(
        hbuf, Wd + (size_t)l * 64, bd + (size_t)l * 2, Vpred, vbuf,
        (l == 2) ? (float*)d_out : nullptr, N);
  }
}
